// ZeroShotJointModel_66735201845586
// MI455X (gfx1250) — matrix operand order repaired and checked by reading, not compiled, not run on hardware
//
#include <hip/hip_runtime.h>
#include <hip/hip_bf16.h>
#include <math.h>

// ---------------------------------------------------------------------------
// MI455X / gfx1250: all heavy GEMMs on the WMMA pipe (v_wmma_f32_16x16x32_bf16),
// f32 -> bf16 once per call.  NN GEMMs: double-buffered LDS B panels, branch-
// free 4x WMMA inner chain.  NT (logits) GEMMs: B panel staged by the Tensor
// Data Mover (tensor_load_to_lds + s_wait_tensorcnt).
// ---------------------------------------------------------------------------

typedef __attribute__((ext_vector_type(16))) __bf16 v16bf;
typedef __attribute__((ext_vector_type(8)))  float  v8f;
typedef __attribute__((ext_vector_type(4)))  unsigned int u32x4;
typedef __attribute__((ext_vector_type(4)))  int i32x4;
typedef __attribute__((ext_vector_type(8)))  int i32x8;

union Frag16 {
    v16bf          v;
    unsigned int   u32[8];
    uint2          u64[4];
    unsigned short u16[16];
};

__device__ __forceinline__ unsigned short f2bf(float f) {
    unsigned int u = __float_as_uint(f);
    unsigned int r = u + 0x7fffu + ((u >> 16) & 1u);   // round-to-nearest-even
    return (unsigned short)(r >> 16);
}

// ------------------------------- f32 -> bf16 -------------------------------
__global__ __launch_bounds__(256) void k_cvt_bf16(const float* __restrict__ s,
                                                  unsigned short* __restrict__ d,
                                                  long n) {
    long i = (long)blockIdx.x * 256 + threadIdx.x;
    long stride = (long)gridDim.x * 256;
    for (; i < n; i += stride) d[i] = f2bf(s[i]);
}

// --------------------------- WMMA GEMM (NN, hot) ---------------------------
// C[M,N] = A[M,K] * B[K,N] + bias (bf16 in, f32 out).  Requires: M%128==0,
// N%64==0, K%32==0.  Block: 8 waves; wave -> 16x64 strip; block -> 128x64.
// Double-buffered LDS B panel; 4 unconditional WMMAs per K-step.
__global__ __launch_bounds__(256)
void k_gemm_nn(const unsigned short* __restrict__ A,
               const unsigned short* __restrict__ Bm,
               const float* __restrict__ bias,
               float* __restrict__ C,
               int M, int N, int K) {
    __shared__ __attribute__((aligned(16))) unsigned short ldsB[2][64 * 36];

    const int tid  = threadIdx.x;
    const int wave = tid >> 5;
    const int lane = tid & 31;
    const int m0   = (blockIdx.y * 8 + wave) * 16;
    const int n0   = blockIdx.x * 64;

    const int  mrow  = m0 + (lane & 15);
    const int  kbA   = (lane < 16) ? 0 : 8;     // ISA 16-bit A-fragment layout
    const int  kbB   = (lane < 16) ? 0 : 8;     // ISA 16-bit B-fragment layout
    const int  nlane = lane & 15;
    const long arow  = (long)mrow * K;

    // Staging map: thread -> (krow, 8 consecutive n).  Coalesced uint4 loads.
    const int  krow = tid >> 3;          // 0..31
    const int  n8   = (tid & 7) * 8;     // 0,8,...,56
    const long gc   = (long)(n0 + n8);

    v8f acc[4];
#pragma unroll
    for (int t = 0; t < 4; ++t)
#pragma unroll
        for (int r = 0; r < 8; ++r) acc[t][r] = 0.f;

    auto loadPanel = [&](int k0) -> uint4 {
        return *(const uint4*)(Bm + (long)(k0 + krow) * N + gc);
    };
    auto storePanel = [&](uint4 vq, unsigned short* buf) {
        union { uint4 q; unsigned short h[8]; } u;
        u.q = vq;
#pragma unroll
        for (int j = 0; j < 8; ++j) buf[(n8 + j) * 36 + krow] = u.h[j];
    };

    uint4 cur = loadPanel(0);
    storePanel(cur, ldsB[0]);
    __syncthreads();

    const int nK = K >> 5;
    for (int kc = 0; kc < nK; ++kc) {
        const int  k0   = kc << 5;
        const bool have = (kc + 1) < nK;
        uint4 nxt;
        if (have) nxt = loadPanel(k0 + 32);   // overlap with WMMA below

        Frag16 af;    // merges into two global_load_b128
#pragma unroll
        for (int i = 0; i < 8; ++i) {
            int kk = (i < 4) ? (kbA + 2 * i) : (16 + kbA + 2 * (i - 4));
            af.u32[i] = *(const unsigned int*)(A + arow + k0 + kk);
        }

        const unsigned short* buf = ldsB[kc & 1];
#pragma unroll
        for (int t = 0; t < 4; ++t) {
            Frag16 bfr;
            const uint2* p = (const uint2*)(buf + (t * 16 + nlane) * 36 + kbB);
            bfr.u64[0] = p[0];
            bfr.u64[1] = p[1];
            bfr.u64[2] = p[4];
            bfr.u64[3] = p[5];
            acc[t] = __builtin_amdgcn_wmma_f32_16x16x32_bf16(
                false, af.v, false, bfr.v, (short)0, acc[t], false, false);
        }

        if (have) storePanel(nxt, ldsB[(kc + 1) & 1]);
        __syncthreads();
    }

    const int mb = (lane < 16) ? 0 : 8;        // ISA f32 C/D layout
#pragma unroll
    for (int t = 0; t < 4; ++t) {
        int col = n0 + t * 16 + nlane;
        float bv = bias[col];
#pragma unroll
        for (int r = 0; r < 8; ++r) {
            int row = m0 + mb + r;
            C[(long)row * N + col] = acc[t][r] + bv;
        }
    }
}

// ------------------------ WMMA GEMM (NT, logits, TDM) ----------------------
// C[M,N] = A[M,K] * B^T (B stored N x K row-major) * (1/temp).
// Requires: M%128==0, N<=64, K%32==0, single N block (grid.x==1).
// B panel (up to 64 rows x 32 bf16, row stride K) is DMA'd into LDS by the
// Tensor Data Mover with LDS padding 32->36 ushorts per row; rows >= N are
// zero-filled by TDM OOB semantics, so the 4x WMMA chain is unconditional.
__global__ __launch_bounds__(256)
void k_gemm_nt(const unsigned short* __restrict__ A,
               const unsigned short* __restrict__ Bm,
               float* __restrict__ C,
               int M, int N, int K,
               long sAz, long sBz, long sCz,
               const float* __restrict__ logtemp) {
    __shared__ __attribute__((aligned(16))) unsigned short ldsB[64 * 36];

    const int z = blockIdx.z;
    A  += (long)z * sAz;
    Bm += (long)z * sBz;
    C  += (long)z * sCz;

    const int tid  = threadIdx.x;
    const int wave = tid >> 5;
    const int lane = tid & 31;
    const int m0   = (blockIdx.y * 8 + wave) * 16;

    const int  mrow  = m0 + (lane & 15);
    const int  kbA   = (lane < 16) ? 0 : 8;
    const int  kbB   = (lane < 16) ? 0 : 8;
    const int  nlane = lane & 15;
    const long arow  = (long)mrow * K;

    v8f acc[4];
#pragma unroll
    for (int t = 0; t < 4; ++t)
#pragma unroll
        for (int r = 0; r < 8; ++r) acc[t][r] = 0.f;

    const unsigned long long gbase = (unsigned long long)(uintptr_t)Bm;
    const unsigned int       lds0  = (unsigned int)(uintptr_t)&ldsB[0];

    for (int k0 = 0; k0 < K; k0 += 32) {
        __syncthreads();                       // all waves done with prev panel
        if (tid < 32) {                        // wave 0 issues the TDM op
            unsigned long long ga = gbase + (unsigned long long)k0 * 2u;
            u32x4 g0;
            g0[0] = 1u;                                  // count=1, user D#
            g0[1] = lds0;                                // lds_addr
            g0[2] = (unsigned int)ga;                    // global_addr[31:0]
            g0[3] = (unsigned int)(ga >> 32) | (2u << 30); // addr[56:32] | type=2
            i32x8 g1;
            g1[0] = (int)((1u << 16)      // data_size = 2 bytes
                        | (1u << 20)      // pad_enable
                        | (3u << 22)      // pad_interval: every 16 dwords (64B row)
                        | (1u << 25));    // pad_amount: 2 dwords -> 36-ushort stride
            g1[1] = (int)(32u << 16);     // tensor_dim0 = 32 (k elems in bounds)
            g1[2] = (int)(((unsigned)N) << 16);  // tensor_dim1 = N (rows>=N -> 0)
            g1[3] = (int)(32u << 16);     // tile_dim0 = 32
            g1[4] = 64;                   // tile_dim1 = 64 LDS rows
            g1[5] = K;                    // tensor_dim0_stride = K elems
            g1[6] = 0;
            g1[7] = 0;
            i32x4 zg = {0, 0, 0, 0};      // groups 2/3 unused (2D tensor)
#if defined(__clang_major__) && (__clang_major__ >= 23)
            i32x8 zg8 = {0, 0, 0, 0, 0, 0, 0, 0};
            __builtin_amdgcn_tensor_load_to_lds(g0, g1, zg, zg, zg8, 0);
#else
            __builtin_amdgcn_tensor_load_to_lds(g0, g1, zg, zg, 0);
#endif
            __builtin_amdgcn_s_wait_tensorcnt(0);
        }
        __syncthreads();                       // panel visible to all waves

        Frag16 af;
#pragma unroll
        for (int i = 0; i < 8; ++i) {
            int kk = (i < 4) ? (kbA + 2 * i) : (16 + kbA + 2 * (i - 4));
            af.u32[i] = *(const unsigned int*)(A + arow + k0 + kk);
        }
#pragma unroll
        for (int t = 0; t < 4; ++t) {
            Frag16 bfr;
            const uint2* p = (const uint2*)(ldsB + (t * 16 + nlane) * 36 + kbB);
            bfr.u64[0] = p[0];
            bfr.u64[1] = p[1];
            bfr.u64[2] = p[4];
            bfr.u64[3] = p[5];
            acc[t] = __builtin_amdgcn_wmma_f32_16x16x32_bf16(
                false, af.v, false, bfr.v, (short)0, acc[t], false, false);
        }
    }

    float tv = expf(*logtemp);
    tv = fminf(fmaxf(tv, 0.01f), 1.0f);
    const float scale = 1.f / tv;
    const int mb = (lane < 16) ? 0 : 8;
#pragma unroll
    for (int t = 0; t < 4; ++t) {
        int col = t * 16 + nlane;
        if (col < N) {
#pragma unroll
            for (int r = 0; r < 8; ++r) {
                int row = m0 + mb + r;
                C[(long)row * N + col] = acc[t][r] * scale;
            }
        }
    }
}

// ----------------------- LayerNorm (+gelu | +L2norm) -----------------------
__global__ __launch_bounds__(256)
void k_ln_ep(const float* __restrict__ X, const float* __restrict__ g,
             const float* __restrict__ be, unsigned short* __restrict__ outb,
             int N, int mode) {
    __shared__ float red[256];
    const long row = blockIdx.x;
    const float* x = X + row * (long)N;
    const int tid = threadIdx.x;

    float s = 0.f, ss = 0.f;
    for (int j = tid; j < N; j += 256) { float v = x[j]; s += v; ss += v * v; }
    red[tid] = s; __syncthreads();
    for (int st = 128; st > 0; st >>= 1) { if (tid < st) red[tid] += red[tid + st]; __syncthreads(); }
    float mean = red[0] / (float)N; __syncthreads();
    red[tid] = ss; __syncthreads();
    for (int st = 128; st > 0; st >>= 1) { if (tid < st) red[tid] += red[tid + st]; __syncthreads(); }
    float var = red[0] / (float)N - mean * mean; __syncthreads();
    float inv = rsqrtf(var + 1e-5f);

    if (mode == 0) {
        for (int j = tid; j < N; j += 256) {
            float y = (x[j] - mean) * inv * g[j] + be[j];
            float ge = 0.5f * y * (1.f + erff(y * 0.70710678118654752f));
            outb[row * (long)N + j] = f2bf(ge);
        }
    } else {
        float nn = 0.f;
        for (int j = tid; j < N; j += 256) {
            float y = (x[j] - mean) * inv * g[j] + be[j];
            nn += y * y;
        }
        red[tid] = nn; __syncthreads();
        for (int st = 128; st > 0; st >>= 1) { if (tid < st) red[tid] += red[tid + st]; __syncthreads(); }
        float id = 1.f / fmaxf(sqrtf(red[0]), 1e-12f);
        for (int j = tid; j < N; j += 256) {
            float y = (x[j] - mean) * inv * g[j] + be[j];
            outb[row * (long)N + j] = f2bf(y * id);
        }
    }
}

// --------------------------- attention-pool finish -------------------------
__global__ __launch_bounds__(256)
void k_pool_finish(const float* __restrict__ zmat, const float* __restrict__ ap_w2,
                   const float* __restrict__ ap_b2,
                   const float* __restrict__ ent_hid, const float* __restrict__ rel_hid,
                   const int* __restrict__ ent_mask, const int* __restrict__ rel_mask,
                   unsigned short* __restrict__ pooled) {
    __shared__ float red[256];
    __shared__ float wts[16];
    const int lab = blockIdx.x;               // 0..767 (512 ent, 256 rel)
    const int tid = threadIdx.x;
    const int t = tid >> 4, sidx = tid & 15;
    const bool isEnt = lab < 512;
    const float* hid = isEnt ? ent_hid + (long)lab * 16 * 768
                             : rel_hid + (long)(lab - 512) * 16 * 768;
    const int* msk = isEnt ? ent_mask + lab * 16 : rel_mask + (lab - 512) * 16;
    const float* zrow = zmat + (long)lab * 16 * 384;

    float p = 0.f;
    for (int j = sidx; j < 384; j += 16) {
        float zv = zrow[t * 384 + j];
        float ge = 0.5f * zv * (1.f + erff(zv * 0.70710678118654752f));
        p += ge * ap_w2[j];
    }
    red[tid] = p; __syncthreads();
    for (int st = 8; st > 0; st >>= 1) { if (sidx < st) red[tid] += red[tid + st]; __syncthreads(); }
    if (tid < 16) {
        float lg = red[tid * 16] + ap_b2[0];
        if (msk[tid] == 0) lg = -INFINITY;
        wts[tid] = lg;
    }
    __syncthreads();
    if (tid == 0) {
        float mx = -INFINITY;
        for (int i = 0; i < 16; ++i) mx = fmaxf(mx, wts[i]);
        float sm = 0.f;
        for (int i = 0; i < 16; ++i) { float e = expf(wts[i] - mx); wts[i] = e; sm += e; }
        float is = 1.f / sm;
        for (int i = 0; i < 16; ++i) wts[i] *= is;
    }
    __syncthreads();
    for (int d = tid; d < 768; d += 256) {
        float acc = 0.f;
        for (int i = 0; i < 16; ++i) acc += wts[i] * hid[i * 768 + d];
        pooled[(long)lab * 768 + d] = f2bf(acc);
    }
}

// ------------------------------ span features ------------------------------
__global__ __launch_bounds__(256)
void k_spanfeat(const float* __restrict__ text, const int* __restrict__ starts,
                const int* __restrict__ ends, unsigned short* __restrict__ feat) {
    const int idx = blockIdx.x;               // b*512 + n
    const int b = idx >> 9;
    const int st = starts[idx];
    const int en = ends[idx];
    const int width = en - st;
    const float* seq = text + (long)b * 2048 * 768;
    unsigned short* out = feat + (long)idx * 2304;
    for (int d = threadIdx.x; d < 768; d += 256) {
        float sv = seq[(long)min(st, 2047) * 768 + d];
        float ev = seq[(long)min(en, 2047) * 768 + d];
        float mx = -INFINITY;
        for (int w = 0; w <= width && w < 16; ++w) {
            int pos = min(st + w, 2047);
            mx = fmaxf(mx, seq[(long)pos * 768 + d]);
        }
        out[d]        = f2bf(sv);
        out[768 + d]  = f2bf(ev);
        out[1536 + d] = f2bf(mx);
    }
}

// --------------------------------- argmax ----------------------------------
__global__ __launch_bounds__(256)
void k_argmax(const float* __restrict__ logits, int* __restrict__ best, int rows) {
    int idx = blockIdx.x * 256 + threadIdx.x;
    if (idx >= rows) return;
    const float* lg = logits + (long)idx * 64;
    int bi = 0; float bv = lg[0];
    for (int i = 1; i < 64; ++i) { float v = lg[i]; if (v > bv) { bv = v; bi = i; } }
    best[idx] = bi;
}

// ------------------------------ rel features -------------------------------
__global__ __launch_bounds__(256)
void k_relfeat(const unsigned short* __restrict__ span_emb,   // (4096,768) bf16
               const unsigned short* __restrict__ labels,     // (768,768) bf16
               const int* __restrict__ best, const int* __restrict__ subj,
               const int* __restrict__ obj, unsigned short* __restrict__ feat) {
    const int idx = blockIdx.x;               // b*256 + p
    const int b = idx >> 8;
    const int sj = subj[idx], ob = obj[idx];
    const long sbase = ((long)b * 512 + sj) * 768;
    const long obase = ((long)b * 512 + ob) * 768;
    const long stb = ((long)b * 64 + best[b * 512 + sj]) * 768;
    const long otb = ((long)b * 64 + best[b * 512 + ob]) * 768;
    unsigned short* out = feat + (long)idx * 3072;
    for (int d = threadIdx.x; d < 768; d += 256) {
        out[d]        = span_emb[sbase + d];
        out[768 + d]  = span_emb[obase + d];
        out[1536 + d] = labels[stb + d];
        out[2304 + d] = labels[otb + d];
    }
}

// ---------------------------------------------------------------------------
extern "C" void kernel_launch(void* const* d_in, const int* in_sizes, int n_in,
                              void* d_out, int out_size, void* d_ws, size_t ws_size,
                              hipStream_t stream) {
    (void)in_sizes; (void)n_in; (void)out_size; (void)ws_size;
    const float* text     = (const float*)d_in[0];
    const float* ent_hid  = (const float*)d_in[1];
    const float* rel_hid  = (const float*)d_in[2];
    const float* ep_w1 = (const float*)d_in[3];   const float* ep_b1 = (const float*)d_in[4];
    const float* ep_g1 = (const float*)d_in[5];   const float* ep_be1 = (const float*)d_in[6];
    const float* ep_w2 = (const float*)d_in[7];   const float* ep_b2 = (const float*)d_in[8];
    const float* ep_g2 = (const float*)d_in[9];   const float* ep_be2 = (const float*)d_in[10];
    const float* rp_w1 = (const float*)d_in[11];  const float* rp_b1 = (const float*)d_in[12];
    const float* rp_g1 = (const float*)d_in[13];  const float* rp_be1 = (const float*)d_in[14];
    const float* rp_w2 = (const float*)d_in[15];  const float* rp_b2 = (const float*)d_in[16];
    const float* rp_g2 = (const float*)d_in[17];  const float* rp_be2 = (const float*)d_in[18];
    const float* lp_w1 = (const float*)d_in[19];  const float* lp_b1 = (const float*)d_in[20];
    const float* lp_g1 = (const float*)d_in[21];  const float* lp_be1 = (const float*)d_in[22];
    const float* lp_w2 = (const float*)d_in[23];  const float* lp_b2 = (const float*)d_in[24];
    const float* lp_g2 = (const float*)d_in[25];  const float* lp_be2 = (const float*)d_in[26];
    const float* ap_w1 = (const float*)d_in[27];  const float* ap_b1 = (const float*)d_in[28];
    const float* ap_w2 = (const float*)d_in[29];  const float* ap_b2 = (const float*)d_in[30];
    const float* logtemp = (const float*)d_in[31];
    const int* ent_mask = (const int*)d_in[32];
    const int* rel_mask = (const int*)d_in[33];
    const int* starts   = (const int*)d_in[34];
    const int* ends     = (const int*)d_in[35];
    const int* psubj    = (const int*)d_in[36];
    const int* pobj     = (const int*)d_in[37];

    float* out_ent = (float*)d_out;                 // (8,512,64)
    float* out_rel = (float*)d_out + 8 * 512 * 64;  // (8,256,32)

    char* base = (char*)d_ws;
    size_t off = 0;
    auto carve = [&](size_t bytes) -> char* {
        char* p = base + off;
        off = (off + bytes + 255) & ~(size_t)255;
        return p;
    };
    unsigned short* w_lp1 = (unsigned short*)carve(589824UL * 2);
    unsigned short* w_lp2 = (unsigned short*)carve(589824UL * 2);
    unsigned short* w_ep1 = (unsigned short*)carve(5308416UL * 2);
    unsigned short* w_ep2 = (unsigned short*)carve(1769472UL * 2);
    unsigned short* w_rp1 = (unsigned short*)carve(9437184UL * 2);
    unsigned short* w_rp2 = (unsigned short*)carve(2359296UL * 2);
    unsigned short* w_ap1 = (unsigned short*)carve(294912UL * 2);
    unsigned short* labhid = (unsigned short*)carve(12288UL * 768 * 2);
    float*          gemmout = (float*)carve(4096UL * 2304 * 4);     // reused
    unsigned short* pooled  = (unsigned short*)carve(768UL * 768 * 2);
    unsigned short* lph     = (unsigned short*)carve(768UL * 768 * 2);
    unsigned short* labels  = (unsigned short*)carve(768UL * 768 * 2);
    unsigned short* spanf   = (unsigned short*)carve(4096UL * 2304 * 2);
    unsigned short* eph     = (unsigned short*)carve(4096UL * 2304 * 2);
    unsigned short* spanemb = (unsigned short*)carve(4096UL * 768 * 2);
    int*            best    = (int*)carve(4096UL * 4);
    unsigned short* relf    = (unsigned short*)carve(2048UL * 3072 * 2);
    unsigned short* rph     = (unsigned short*)carve(2048UL * 3072 * 2);
    unsigned short* relproj = (unsigned short*)carve(2048UL * 768 * 2);

    auto cvt = [&](const float* s, unsigned short* d, long n) {
        int g = (int)((n + 255) / 256); if (g > 4096) g = 4096;
        k_cvt_bf16<<<g, 256, 0, stream>>>(s, d, n);
    };
    cvt(lp_w1, w_lp1, 589824);      cvt(lp_w2, w_lp2, 589824);
    cvt(ep_w1, w_ep1, 5308416);     cvt(ep_w2, w_ep2, 1769472);
    cvt(rp_w1, w_rp1, 9437184);     cvt(rp_w2, w_rp2, 2359296);
    cvt(ap_w1, w_ap1, 294912);
    cvt(ent_hid, labhid, 6291456);                       // rows 0..8191
    cvt(rel_hid, labhid + 8192UL * 768, 3145728);        // rows 8192..12287

    // --- label attention pool: z = labhid @ ap_w1 + b1  (12288x384) --------
    k_gemm_nn<<<dim3(6, 96, 1), 256, 0, stream>>>(labhid, w_ap1, ap_b1, gemmout,
                                                  12288, 384, 768);
    k_pool_finish<<<768, 256, 0, stream>>>(gemmout, ap_w2, ap_b2, ent_hid,
                                           rel_hid, ent_mask, rel_mask, pooled);

    // --- label MLP ('lp') --------------------------------------------------
    k_gemm_nn<<<dim3(12, 6, 1), 256, 0, stream>>>(pooled, w_lp1, lp_b1, gemmout,
                                                  768, 768, 768);
    k_ln_ep<<<768, 256, 0, stream>>>(gemmout, lp_g1, lp_be1, lph, 768, 0);
    k_gemm_nn<<<dim3(12, 6, 1), 256, 0, stream>>>(lph, w_lp2, lp_b2, gemmout,
                                                  768, 768, 768);
    k_ln_ep<<<768, 256, 0, stream>>>(gemmout, lp_g2, lp_be2, labels, 768, 1);

    // --- span features -----------------------------------------------------
    k_spanfeat<<<4096, 256, 0, stream>>>(text, starts, ends, spanf);

    // --- span MLP ('ep') ---------------------------------------------------
    k_gemm_nn<<<dim3(36, 32, 1), 256, 0, stream>>>(spanf, w_ep1, ep_b1, gemmout,
                                                   4096, 2304, 2304);
    k_ln_ep<<<4096, 256, 0, stream>>>(gemmout, ep_g1, ep_be1, eph, 2304, 0);
    k_gemm_nn<<<dim3(12, 32, 1), 256, 0, stream>>>(eph, w_ep2, ep_b2, gemmout,
                                                   4096, 768, 2304);
    k_ln_ep<<<4096, 256, 0, stream>>>(gemmout, ep_g2, ep_be2, spanemb, 768, 1);

    // --- entity logits (TDM-staged NT GEMM) --------------------------------
    k_gemm_nt<<<dim3(1, 4, 8), 256, 0, stream>>>(
        spanemb, labels, out_ent, 512, 64, 768,
        512L * 768, 64L * 768, 512L * 64, logtemp);

    k_argmax<<<16, 256, 0, stream>>>(out_ent, best, 4096);
    k_relfeat<<<2048, 256, 0, stream>>>(spanemb, labels, best, psubj, pobj, relf);

    // --- relation MLP ('rp') -----------------------------------------------
    k_gemm_nn<<<dim3(48, 16, 1), 256, 0, stream>>>(relf, w_rp1, rp_b1, gemmout,
                                                   2048, 3072, 3072);
    k_ln_ep<<<2048, 256, 0, stream>>>(gemmout, rp_g1, rp_be1, rph, 3072, 0);
    k_gemm_nn<<<dim3(12, 16, 1), 256, 0, stream>>>(rph, w_rp2, rp_b2, gemmout,
                                                   2048, 768, 3072);
    k_ln_ep<<<2048, 256, 0, stream>>>(gemmout, rp_g2, rp_be2, relproj, 768, 1);

    // --- relation logits (TDM-staged NT GEMM) ------------------------------
    k_gemm_nt<<<dim3(1, 2, 8), 256, 0, stream>>>(
        relproj, labels + 512UL * 768, out_rel, 256, 32, 768,
        256L * 768, 32L * 768, 256L * 32, logtemp);
}
